// Embedding_50706383896913
// MI455X (gfx1250) — matrix operand order repaired and checked by reading, not compiled, not run on hardware
//
#include <hip/hip_runtime.h>

// ---------------------------------------------------------------------------
// E[b,g,z] = sum_l Wc[g,l] * M[b,l,z] + Wp[g,z]
// Split-bf16 (bf16x3) GEMM on v_wmma_f32_16x16x32_bf16.
// Memory-bound design (M = 339 MB streamed once; HBM floor ~15 us @ 23.3 TB/s).
// Software-pipelined: next M panel's global loads are in flight during WMMA.
// ---------------------------------------------------------------------------

typedef __attribute__((ext_vector_type(16))) __bf16 v16bf;
typedef __attribute__((ext_vector_type(8)))  float  v8f;

#define L_DIM   2586
#define ZETA    128
#define GAMMA   128
#define BATCH   256
#define KSTEP   32
#define KSTEPS  81                 // ceil(2586/32); padded K range = 2592
#define SLOT_US 24                 // 48 bytes per lane slot (32B frag + 16B pad)
#define ZT_US   (32 * SLOT_US)     // 768 ushorts per 16-col z-tile
#define PANEL_US (8 * ZT_US)       // 6144 ushorts per hi (or lo) panel

struct Frag32 { uint4 a, b; };     // 32 bytes == one v16bf fragment per lane

static __device__ __forceinline__ v16bf frag_cast(const Frag32& f) {
    return __builtin_bit_cast(v16bf, f);
}

// Round-to-nearest-even fp32 -> bf16 split: x ~= hi + lo
static __device__ __forceinline__ void split2(float x, unsigned short& hi,
                                              unsigned short& lo) {
    unsigned int u  = __float_as_uint(x);
    unsigned int rh = u + 0x7FFFu + ((u >> 16) & 1u);
    unsigned short h = (unsigned short)(rh >> 16);
    float hf = __uint_as_float(((unsigned int)h) << 16);
    float l  = x - hf;
    unsigned int ul = __float_as_uint(l);
    unsigned int rl = ul + 0x7FFFu + ((ul >> 16) & 1u);
    hi = h;
    lo = (unsigned short)(rl >> 16);
}

// ---------------------------------------------------------------------------
// Prep: Wc (128 x 2586 fp32) -> pre-swizzled A fragments (hi/lo bf16).
// Layout: [gt(8)][ks(81)][lane(32)][h(16)] ushort, in this file's
// contiguous-K A-fragment order (same K order as the B panel):
//   lanes 0-15 : M=lane,    K=h       (K 0-15)
//   lanes 16-31: M=lane-16, K=h+16    (K 16-31)
// ---------------------------------------------------------------------------
__global__ void prep_wc_kernel(const float* __restrict__ Wc,
                               unsigned short* __restrict__ a_hi,
                               unsigned short* __restrict__ a_lo) {
    int t = blockIdx.x * blockDim.x + threadIdx.x;
    if (t >= 8 * KSTEPS * 32) return;
    int lane = t & 31;
    int ks   = (t >> 5) % KSTEPS;
    int gt   = t / (KSTEPS * 32);
    int m    = lane & 15;
    int hl   = lane >> 4;          // lane-half selector

    unsigned short hs[16], ls[16];
#pragma unroll
    for (int h = 0; h < 16; ++h) {
        int K = h + hl * 16;
        int l = ks * KSTEP + K;
        float v = (l < L_DIM) ? Wc[(size_t)(gt * 16 + m) * L_DIM + l] : 0.0f;
        split2(v, hs[h], ls[h]);
    }
    unsigned int wh[8], wl[8];
#pragma unroll
    for (int j = 0; j < 8; ++j) {
        wh[j] = (unsigned int)hs[2*j] | ((unsigned int)hs[2*j+1] << 16);
        wl[j] = (unsigned int)ls[2*j] | ((unsigned int)ls[2*j+1] << 16);
    }
    size_t base = (size_t)t * 16;
    *(uint4*)&a_hi[base]     = make_uint4(wh[0], wh[1], wh[2], wh[3]);
    *(uint4*)&a_hi[base + 8] = make_uint4(wh[4], wh[5], wh[6], wh[7]);
    *(uint4*)&a_lo[base]     = make_uint4(wl[0], wl[1], wl[2], wl[3]);
    *(uint4*)&a_lo[base + 8] = make_uint4(wl[4], wl[5], wl[6], wl[7]);
}

// Load one thread's 16-element column slice of the 32x128 M panel for step ks.
// Wave-uniform branch: guarded path only ever taken for ks==80, khalf==1.
static __device__ __forceinline__ void load_panel(const float* __restrict__ Mb,
                                                  int ks, int khalf, int z,
                                                  float v[16]) {
    int kbase = ks * KSTEP + khalf * 16;
    const float* src = Mb + (size_t)kbase * ZETA + z;
    if (kbase + 16 <= L_DIM) {
#pragma unroll
        for (int k = 0; k < 16; ++k) v[k] = src[k * ZETA];
    } else {
#pragma unroll
        for (int k = 0; k < 16; ++k)
            v[k] = (kbase + k < L_DIM) ? src[k * ZETA] : 0.0f;
    }
}

// ---------------------------------------------------------------------------
// Main: one workgroup per batch, 8 wave32s. Wave w owns g-rows [16w,16w+16)
// x all 128 z columns (8 accum tiles). M panel (32 x 128 fp32) staged through
// LDS as bf16 hi/lo in per-lane B-fragment order (contiguous K, as A):
//   B 32x16 bf16: lanes 0-15: N=lane, K=h; lanes 16-31: N=lane-16, K=16+h.
// Pipeline per step: store panel(ks) -> barrier -> issue loads(ks+1) +
// A frags + 24 WMMA -> barrier.
// ---------------------------------------------------------------------------
__global__ void __launch_bounds__(256)
embed_gemm_kernel(const float* __restrict__ M,
                  const float* __restrict__ Wp,
                  const unsigned short* __restrict__ a_hi,
                  const unsigned short* __restrict__ a_lo,
                  float* __restrict__ out) {
    __shared__ __align__(16) unsigned short smem[2 * PANEL_US];
    unsigned short* Bhi = smem;
    unsigned short* Blo = smem + PANEL_US;

    const int b    = blockIdx.x;
    const int t    = threadIdx.x;
    const int lane = t & 31;
    const int w    = t >> 5;        // wave id = g-tile id
    // staging coordinates: thread owns column z, 16 consecutive K rows
    const int z     = t & 127;
    const int khalf = t >> 7;       // 0: K 0-15, 1: K 16-31  (wave-uniform)
    const int s_off = (z >> 4) * ZT_US + ((z & 15) + (khalf << 4)) * SLOT_US;

    const float* Mb = M + (size_t)b * L_DIM * ZETA;

    v8f acc[8];
#pragma unroll
    for (int i = 0; i < 8; ++i)
        acc[i] = (v8f){0.f, 0.f, 0.f, 0.f, 0.f, 0.f, 0.f, 0.f};

    // ---- pipeline prologue: loads for ks = 0 ----
    float vcur[16];
    load_panel(Mb, 0, khalf, z, vcur);

    for (int ks = 0; ks < KSTEPS; ++ks) {
        // ---- convert current panel slice to bf16 hi/lo, store to LDS ----
        {
            unsigned short hs[16], ls[16];
#pragma unroll
            for (int k = 0; k < 16; ++k) split2(vcur[k], hs[k], ls[k]);
            unsigned int wh[8], wl[8];
#pragma unroll
            for (int j = 0; j < 8; ++j) {
                wh[j] = (unsigned int)hs[2*j] | ((unsigned int)hs[2*j+1] << 16);
                wl[j] = (unsigned int)ls[2*j] | ((unsigned int)ls[2*j+1] << 16);
            }
            *(uint4*)&Bhi[s_off]     = make_uint4(wh[0], wh[1], wh[2], wh[3]);
            *(uint4*)&Bhi[s_off + 8] = make_uint4(wh[4], wh[5], wh[6], wh[7]);
            *(uint4*)&Blo[s_off]     = make_uint4(wl[0], wl[1], wl[2], wl[3]);
            *(uint4*)&Blo[s_off + 8] = make_uint4(wl[4], wl[5], wl[6], wl[7]);
        }
        __syncthreads();

        // ---- issue next panel's global loads (latency hidden by WMMA) ----
        float vnext[16];
        if (ks + 1 < KSTEPS) load_panel(Mb, ks + 1, khalf, z, vnext);

        // ---- A fragments (pre-swizzled, L2-resident) ----
        const size_t abase = ((size_t)(w * KSTEPS + ks) * 32 + lane) * 16;
        const uint4* pah = (const uint4*)&a_hi[abase];
        const uint4* pal = (const uint4*)&a_lo[abase];
        Frag32 fah = { pah[0], pah[1] };
        Frag32 fal = { pal[0], pal[1] };
        v16bf Ah = frag_cast(fah);
        v16bf Al = frag_cast(fal);

        // ---- 8 z-tiles x 3 WMMA (bf16x3 split product) ----
#pragma unroll
        for (int zt = 0; zt < 8; ++zt) {
            const int boff = zt * ZT_US + lane * SLOT_US;
            const uint4* pbh = (const uint4*)&Bhi[boff];
            const uint4* pbl = (const uint4*)&Blo[boff];
            Frag32 fbh = { pbh[0], pbh[1] };
            Frag32 fbl = { pbl[0], pbl[1] };
            v16bf Bh = frag_cast(fbh);
            v16bf Bl = frag_cast(fbl);
            acc[zt] = __builtin_amdgcn_wmma_f32_16x16x32_bf16(
                false, Ah, false, Bh, (short)0, acc[zt], false, false);
            acc[zt] = __builtin_amdgcn_wmma_f32_16x16x32_bf16(
                false, Ah, false, Bl, (short)0, acc[zt], false, false);
            acc[zt] = __builtin_amdgcn_wmma_f32_16x16x32_bf16(
                false, Al, false, Bh, (short)0, acc[zt], false, false);
        }
        __syncthreads();

#pragma unroll
        for (int k = 0; k < 16; ++k) vcur[k] = vnext[k];
    }

    // ---- epilogue: C/D layout -> global, with Wp bias ----
    // VGPR r: lanes 0-15 -> M=r, N=lane; lanes 16-31 -> M=8+r, N=lane-16
    const int n  = lane & 15;
    const int m0 = (lane >> 4) * 8;
#pragma unroll
    for (int zt = 0; zt < 8; ++zt) {
#pragma unroll
        for (int r = 0; r < 8; ++r) {
            int g  = w * 16 + m0 + r;
            int zz = zt * 16 + n;
            out[((size_t)b * GAMMA + g) * ZETA + zz] =
                acc[zt][r] + Wp[g * ZETA + zz];
        }
    }
}

// ---------------------------------------------------------------------------
extern "C" void kernel_launch(void* const* d_in, const int* in_sizes, int n_in,
                              void* d_out, int out_size, void* d_ws, size_t ws_size,
                              hipStream_t stream) {
    (void)in_sizes; (void)n_in; (void)out_size; (void)ws_size;
    const float* M  = (const float*)d_in[0];   // (256, 2586, 128) fp32
    const float* Wc = (const float*)d_in[1];   // (128, 2586) fp32
    const float* Wp = (const float*)d_in[2];   // (128, 128) fp32
    float* out = (float*)d_out;                // (256, 128, 128) fp32

    unsigned short* a_hi = (unsigned short*)d_ws;                  // 331776 ushorts
    unsigned short* a_lo = a_hi + (size_t)8 * KSTEPS * 32 * 16;    // another 331776

    const int prep_threads = 8 * KSTEPS * 32;  // 20736
    prep_wc_kernel<<<(prep_threads + 255) / 256, 256, 0, stream>>>(Wc, a_hi, a_lo);
    embed_gemm_kernel<<<BATCH, 256, 0, stream>>>(M, Wp, a_hi, a_lo, out);
}
